// LSTMEncoderDecoder_36412732736093
// MI455X (gfx1250) — compile-verified, matrix operand order checked by reading, not run on hardware
//
#include <hip/hip_runtime.h>
#include <hip/hip_bf16.h>
#include <stdint.h>

// ---------------------------------------------------------------------------
// LSTM encoder-decoder on MI455X (gfx1250).
// Per-step kernels (launch boundary = device-wide sync for the recurrence),
// bf16 WMMA 16x16x32 with f32 accumulation, decoder dense projection folded
// into the recurrent weight matrix: W_eff = Whh_dec + Wih_dec*Wd.
// ---------------------------------------------------------------------------

typedef __attribute__((ext_vector_type(16))) __bf16 v16bf;
typedef __attribute__((ext_vector_type(8)))  float  v8f;

#define Bsz 256
#define Tsz 1024
#define Fsz 64
#define Esz 256
#define NG  1024  // 4*E gate columns

union FragBF { v16bf v; uint32_t u[8]; };

// 16-bit A-matrix 16x32 K mapping (ISA 7.12.2): VGPR v holds a K pair.
__device__ __forceinline__ int kpair(int v, int hi) {
  return 2 * (v & 3) + 8 * hi + 16 * (v >> 2);
}

__device__ __forceinline__ float sigmoidf_(float x) {
  return 1.0f / (1.0f + __expf(-x));
}

// ---------------------------------------------------------------------------
// One LSTM time step (+ optional projection blocks for the decoder).
//   gate blocks  : blockIdx.x <  nGateBlocks  (32 blocks, 8 waves each;
//                  block = 16 rows x 128 gate-cols, wave = 16x16 per gate)
//   proj blocks  : blockIdx.x >= nGateBlocks  (project h_in -> out[t_out])
// ---------------------------------------------------------------------------
__global__ void lstm_step_kernel(
    const float*  __restrict__ x,      // (B,T,F) fp32, null for decoder
    const __bf16* __restrict__ W,      // [NG][kdim] bf16, k-contiguous
    const float*  __restrict__ bias,   // [NG] fp32 (bih+bhh [+Wih*bd])
    const __bf16* __restrict__ h_in,   // [B][E] bf16
    __bf16*       __restrict__ h_out,  // [B][E] bf16
    float*        __restrict__ c,      // [B][E] fp32, updated in place
    const __bf16* __restrict__ projW,  // [F][E] bf16 (== Wd cast)
    const float*  __restrict__ bproj,  // [F] fp32 (bd)
    float*        __restrict__ out,    // (T,B,F) fp32 time-major
    int t, int t_out, int kdim, int nGateBlocks)
{
  const int lane = threadIdx.x & 31;
  const int wave = threadIdx.x >> 5;
  const int hi   = lane >> 4;
  const int ln   = lane & 15;

  if ((int)blockIdx.x < nGateBlocks) {
    // ---------------- recurrent gate GEMM + cell update ----------------
    const int mt  = blockIdx.x >> 1;          // 0..15 row tile (16 rows)
    const int nb  = blockIdx.x & 1;           // 0..1  col half (128 cols)
    const int m0  = mt * 16;
    const int n0  = nb * 128 + wave * 16;     // gate-col base within E
    const int row = m0 + ln;                  // A row (batch index) per lane

    v8f acc[4];
#pragma unroll
    for (int q = 0; q < 4; ++q) {
      const float bq = bias[n0 + ln + q * Esz];
#pragma unroll
      for (int r = 0; r < 8; ++r) acc[q][r] = bq;
    }

    const int nk = kdim / 32;
    for (int kk = 0; kk < nk; ++kk) {
      const int kb = kk * 32;
      FragBF a;
      if (x && kb < Fsz) {
        // A-fragment from x_t (fp32 -> bf16 on the fly)
        const float* xr = x + (size_t)row * (Tsz * Fsz) + (size_t)t * Fsz + kb;
#pragma unroll
        for (int v = 0; v < 8; ++v) {
          const int kp = kpair(v, hi);
          union { uint32_t u; __bf16 h[2]; } pk;
          pk.h[0] = (__bf16)xr[kp];
          pk.h[1] = (__bf16)xr[kp + 1];
          a.u[v] = pk.u;
        }
      } else {
        // A-fragment from h_in (bf16 pairs, 32-bit loads)
        const __bf16* hr = h_in + (size_t)row * Esz + (kb - (x ? Fsz : 0));
#pragma unroll
        for (int v = 0; v < 8; ++v)
          a.u[v] = *(const uint32_t*)(hr + kpair(v, hi));
      }
      // 4 gate tiles reuse the same A-fragment
#pragma unroll
      for (int q = 0; q < 4; ++q) {
        const __bf16* wr = W + (size_t)(n0 + ln + q * Esz) * kdim + kb;
        FragBF bf;
#pragma unroll
        for (int v = 0; v < 8; ++v)
          bf.u[v] = *(const uint32_t*)(wr + kpair(v, hi));
        acc[q] = __builtin_amdgcn_wmma_f32_16x16x32_bf16(
            false, a.v, false, bf.v, (short)0, acc[q], false, false);
      }
    }

    // fused LSTM cell: i,f,g,o -> c,h   (lane owns col ncol, rows r+8*hi)
    const int ncol = n0 + ln;
#pragma unroll
    for (int r = 0; r < 8; ++r) {
      const int b = m0 + r + 8 * hi;
      const float iv = sigmoidf_(acc[0][r]);
      const float fv = sigmoidf_(acc[1][r]);
      const float gv = tanhf(acc[2][r]);
      const float ov = sigmoidf_(acc[3][r]);
      float cc = c[(size_t)b * Esz + ncol];
      cc = fv * cc + iv * gv;
      c[(size_t)b * Esz + ncol] = cc;
      h_out[(size_t)b * Esz + ncol] = (__bf16)(ov * tanhf(cc));
    }
  } else {
    // ---------------- dense projection: out[t_out] = h_in*Wd^T + bd ----
    const int p    = blockIdx.x - nGateBlocks;  // 0..7
    const int tile = p * 8 + wave;              // 0..63
    const int mt   = tile >> 2;                 // 0..15
    const int nt   = tile & 3;                  // 0..3
    const int m0   = mt * 16;
    const int n0   = nt * 16;
    const int row  = m0 + ln;

    v8f acc;
    {
      const float bq = bproj[n0 + ln];
#pragma unroll
      for (int r = 0; r < 8; ++r) acc[r] = bq;
    }
    for (int kk = 0; kk < Esz / 32; ++kk) {
      const int kb = kk * 32;
      const __bf16* hr = h_in + (size_t)row * Esz + kb;
      const __bf16* wr = projW + (size_t)(n0 + ln) * Esz + kb;
      FragBF a, bf;
#pragma unroll
      for (int v = 0; v < 8; ++v) {
        const int kp = kpair(v, hi);
        a.u[v]  = *(const uint32_t*)(hr + kp);
        bf.u[v] = *(const uint32_t*)(wr + kp);
      }
      acc = __builtin_amdgcn_wmma_f32_16x16x32_bf16(
          false, a.v, false, bf.v, (short)0, acc, false, false);
    }
    float* orow = out + (size_t)t_out * (Bsz * Fsz);
    const int fcol = n0 + ln;
#pragma unroll
    for (int r = 0; r < 8; ++r) {
      const int b = m0 + r + 8 * hi;
      orow[(size_t)b * Fsz + fcol] = acc[r];
    }
  }
}

// ---------------------------------------------------------------------------
// One-shot setup: pack weights to bf16 [n][k] layout, fold decoder dense
// projection into the recurrent weights, fold biases, zero h0/c0.
// ---------------------------------------------------------------------------
__global__ void setup_kernel(
    const float* __restrict__ enc_Wih, const float* __restrict__ enc_Whh,
    const float* __restrict__ enc_bih, const float* __restrict__ enc_bhh,
    const float* __restrict__ dec_Wih, const float* __restrict__ dec_Whh,
    const float* __restrict__ dec_bih, const float* __restrict__ dec_bhh,
    const float* __restrict__ Wd,      const float* __restrict__ bd,
    __bf16* __restrict__ encW,   // [NG][320]
    __bf16* __restrict__ decW,   // [NG][256]
    __bf16* __restrict__ projW,  // [64][256]
    float*  __restrict__ encB,   // [NG]
    float*  __restrict__ decB,   // [NG]
    __bf16* __restrict__ h0,     // [B][E]
    float*  __restrict__ c0)     // [B][E]
{
  const int idx = blockIdx.x * blockDim.x + threadIdx.x;

  // encoder combined weight [n][k], k = [x(64) | h(256)]
  if (idx < NG * 320) {
    const int n = idx / 320, k = idx % 320;
    encW[idx] = (__bf16)(k < Fsz ? enc_Wih[n * Fsz + k]
                                 : enc_Whh[n * Esz + (k - Fsz)]);
  }
  // decoder fused weight: W_eff[n][e] = Whh[n][e] + sum_f Wih[n][f]*Wd[f][e]
  if (idx < NG * Esz) {
    const int n = idx / Esz, e = idx % Esz;
    float s = dec_Whh[n * Esz + e];
    for (int f = 0; f < Fsz; ++f)
      s += dec_Wih[n * Fsz + f] * Wd[f * Esz + e];
    decW[idx] = (__bf16)s;
  }
  // projection weight: [n=f][k=e] == Wd row-major, cast
  if (idx < Fsz * Esz) projW[idx] = (__bf16)Wd[idx];
  // biases
  if (idx < NG) {
    encB[idx] = enc_bih[idx] + enc_bhh[idx];
    float s = dec_bih[idx] + dec_bhh[idx];
    for (int f = 0; f < Fsz; ++f) s += dec_Wih[idx * Fsz + f] * bd[f];
    decB[idx] = s;
  }
  // zero initial state
  if (idx < Bsz * Esz) { h0[idx] = (__bf16)0.0f; c0[idx] = 0.0f; }
}

// ---------------------------------------------------------------------------
extern "C" void kernel_launch(void* const* d_in, const int* in_sizes, int n_in,
                              void* d_out, int out_size, void* d_ws, size_t ws_size,
                              hipStream_t stream) {
  const float* x        = (const float*)d_in[0];
  const float* enc_Wih  = (const float*)d_in[1];
  const float* enc_Whh  = (const float*)d_in[2];
  const float* enc_bih  = (const float*)d_in[3];
  const float* enc_bhh  = (const float*)d_in[4];
  const float* dec_Wih  = (const float*)d_in[5];
  const float* dec_Whh  = (const float*)d_in[6];
  const float* dec_bih  = (const float*)d_in[7];
  const float* dec_bhh  = (const float*)d_in[8];
  const float* Wd       = (const float*)d_in[9];
  const float* bd       = (const float*)d_in[10];
  float* out            = (float*)d_out;

  // workspace layout (bytes)
  char* ws = (char*)d_ws;
  size_t off = 0;
  auto alloc = [&](size_t bytes) { char* p = ws + off; off = (off + bytes + 255) & ~(size_t)255; return p; };
  __bf16* encW  = (__bf16*)alloc((size_t)NG * 320 * 2);
  __bf16* decW  = (__bf16*)alloc((size_t)NG * Esz * 2);
  __bf16* projW = (__bf16*)alloc((size_t)Fsz * Esz * 2);
  float*  encB  = (float*)alloc((size_t)NG * 4);
  float*  decB  = (float*)alloc((size_t)NG * 4);
  __bf16* hA    = (__bf16*)alloc((size_t)Bsz * Esz * 2);
  __bf16* hB    = (__bf16*)alloc((size_t)Bsz * Esz * 2);
  float*  cbuf  = (float*)alloc((size_t)Bsz * Esz * 4);
  if (off > ws_size) return;  // workspace too small; bail deterministically

  // one-shot weight packing / folding / state init
  {
    const int nthreads = NG * 320;
    setup_kernel<<<(nthreads + 255) / 256, 256, 0, stream>>>(
        enc_Wih, enc_Whh, enc_bih, enc_bhh, dec_Wih, dec_Whh, dec_bih,
        dec_bhh, Wd, bd, encW, decW, projW, encB, decB, hA, cbuf);
  }

  __bf16* h_cur = hA;
  __bf16* h_nxt = hB;

  // ---- encoder: 1024 recurrent steps (kdim = 320: x|h) ----
  for (int t = 0; t < Tsz; ++t) {
    lstm_step_kernel<<<dim3(32), dim3(256), 0, stream>>>(
        x, encW, encB, h_cur, h_nxt, cbuf,
        nullptr, nullptr, nullptr, t, 0, 320, 32);
    __bf16* tmp = h_cur; h_cur = h_nxt; h_nxt = tmp;
  }

  // ---- decoder: 1023 fused steps; 8 extra blocks project prev h ----
  for (int s = 1; s < Tsz; ++s) {
    lstm_step_kernel<<<dim3(40), dim3(256), 0, stream>>>(
        nullptr, decW, decB, h_cur, h_nxt, cbuf,
        projW, bd, out, 0, s - 1, Esz, 32);
    __bf16* tmp = h_cur; h_cur = h_nxt; h_nxt = tmp;
  }

  // ---- final projection: out[T-1] = h_{T-1} * Wd^T + bd ----
  lstm_step_kernel<<<dim3(8), dim3(256), 0, stream>>>(
      nullptr, decW, decB, h_cur, h_nxt, cbuf,
      projW, bd, out, 0, Tsz - 1, Esz, 0);
}
